// BinarySparseCellNet_3659312136776
// MI455X (gfx1250) — matrix operand order repaired and checked by reading, not compiled, not run on hardware
//
#include <hip/hip_runtime.h>

// ---------------------------------------------------------------------------
// CDNA5 (gfx1250) implementation of BinarySparseCellNet.
// Core compute: one generic implicit-GEMM conv kernel using
// V_WMMA_F32_16X16X32_F16 (wave32). Per wave: M-tile = 32 voxels (2 A-frags),
// K-chunk = 32 f16 of flattened (tap, cin), up to 4 N-tiles of 16 output
// channels (block-shared B tile). Masked-BN + ReLU fused into A staging;
// mask-mult + residual fused into the C store; channel stride/offset params
// implement skip-concat in place. Weight tile for the next K-chunk is
// prefetched (global_prefetch_b8) while current WMMAs execute.
// ---------------------------------------------------------------------------

typedef __attribute__((ext_vector_type(16))) _Float16 v16h;
typedef __attribute__((ext_vector_type(8)))  _Float16 v8h;
typedef __attribute__((ext_vector_type(8)))  float    v8f;

#define CONV_WAVES 4
#define MSUB 2   // 16-voxel sub-tiles per wave

struct ConvArgs {
    const float* x;        // input grid  [ID,IH,IW, in_cstride], channel offset in_coffset
    const float* w;        // weights [T, Cin, Cout] (taps major) (+tap base for up mode)
    const float* mask_in;  // mask at input resolution (used with bn), or null
    const float* mask_out; // multiply output by mask, or null
    const float* bnscale;  // per-in-channel scale (null => raw input, no ReLU)
    const float* bnshift;
    const float* residual; // add at store; layout [vox][Cout]; or null
    float* y;              // output [OD,OH,OW, out_cstride] at out_coffset
    int ID, IH, IW, Cin, in_cstride, in_coffset;
    int OD, OH, OW, Cout, out_cstride, out_coffset;
    int KD, KH, KW;        // kernel taps
    int SD, SH, SW;        // strides
    int PD, PH, PW;        // pads
    int up_mode;           // 1 => conv_transpose with stride==kernel; tap = blockIdx.z
};

__global__ __launch_bounds__(32 * CONV_WAVES)
void conv_wmma_kernel(ConvArgs a) {
    __shared__ __align__(16) _Float16 As[CONV_WAVES][MSUB][16][32]; // per-wave A tiles
    __shared__ __align__(32) _Float16 Bt[4][16][32];                // block-shared B (nt x n x k)

    const int lane  = threadIdx.x;          // 0..31
    const int wv    = threadIdx.y;          // 0..3
    const int flat  = wv * 32 + lane;
    const int mbase = (blockIdx.x * CONV_WAVES + wv) * (16 * MSUB);
    const int g     = lane >> 4;            // lane group (0/1)
    const int nn    = lane & 15;

    const int NT   = (a.Cout + 15) >> 4;                               // <= 4
    const int Mtot = a.up_mode ? a.ID * a.IH * a.IW : a.OD * a.OH * a.OW;
    const int Ktot = a.up_mode ? a.Cin : a.KD * a.KH * a.KW * a.Cin;

    int wbase = 0, utd = 0, uth = 0, utw = 0;
    if (a.up_mode) {
        int t = blockIdx.z;
        utd = t / (a.SH * a.SW);
        uth = (t / a.SW) % a.SH;
        utw = t % a.SW;
        wbase = ((utd * a.KH + uth) * a.KW + utw) * a.Cin * a.Cout;
    }

    v8f acc[MSUB][4] = {};

    for (int kc = 0; kc < Ktot; kc += 32) {
        // ----- stage A (per wave): lane = k, loop j = voxel row -----
        {
            int kk = kc + lane;
            bool kok = kk < Ktot;
            int cin = 0, td = 0, th = 0, tw = 0;
            float scl = 1.f, sft = 0.f;
            if (kok) {
                if (a.up_mode) { cin = kk; }
                else {
                    int tap = kk / a.Cin; cin = kk - tap * a.Cin;
                    td = tap / (a.KH * a.KW);
                    th = (tap / a.KW) % a.KH;
                    tw = tap % a.KW;
                }
                if (a.bnscale) { scl = a.bnscale[cin]; sft = a.bnshift[cin]; }
            }
            #pragma unroll
            for (int j = 0; j < 16 * MSUB; ++j) {
                float val = 0.f;
                int vox = mbase + j;
                if (kok && vox < Mtot) {
                    long vidx = -1;
                    if (a.up_mode) {
                        vidx = vox;
                    } else {
                        int od = vox / (a.OH * a.OW);
                        int rem = vox - od * (a.OH * a.OW);
                        int oh = rem / a.OW;
                        int ow = rem - oh * a.OW;
                        int id = od * a.SD - a.PD + td;
                        int ih = oh * a.SH - a.PH + th;
                        int iw = ow * a.SW - a.PW + tw;
                        if ((unsigned)id < (unsigned)a.ID &&
                            (unsigned)ih < (unsigned)a.IH &&
                            (unsigned)iw < (unsigned)a.IW)
                            vidx = ((long)id * a.IH + ih) * a.IW + iw;
                    }
                    if (vidx >= 0) {
                        float xv = a.x[vidx * a.in_cstride + a.in_coffset + cin];
                        if (a.bnscale) {   // fused bn_act: relu(x*s+b) * mask
                            float mk = a.mask_in ? a.mask_in[vidx] : 1.f;
                            float yv = xv * scl + sft;
                            val = (mk != 0.f) ? fmaxf(yv, 0.f) : 0.f;
                        } else val = xv;
                    }
                }
                As[wv][j >> 4][j & 15][lane] = (_Float16)val;
            }
        }
        // ----- stage B (block shared): NT*512 halves over 128 threads -----
        for (int idx = flat; idx < NT * 512; idx += 128) {
            int nt = idx >> 9;
            int r  = idx & 511;
            int n  = r >> 5;
            int k  = r & 31;
            int kk = kc + k;
            int col = nt * 16 + n;
            float wvv = (kk < Ktot && col < a.Cout)
                      ? a.w[wbase + (long)kk * a.Cout + col] : 0.f;
            Bt[nt][n][k] = (_Float16)wvv;
        }
        // prefetch next K-chunk's weight rows while WMMAs run (global_prefetch_b8)
        if (kc + 32 < Ktot)
            __builtin_prefetch(&a.w[wbase + (long)(kc + 32 + lane) * a.Cout], 0, 1);
        __syncthreads();

        // ----- gather fragments per CDNA5 layouts, issue WMMA -----
        // A (16-bit 16x32): lane m=lane%16, halves 0-7 -> K = 8*g + h,
        //                   halves 8-15 -> K = 16 + 8*g + (h-8)
        union { v16h v; v8h h[2]; } ua[MSUB];
        #pragma unroll
        for (int mi = 0; mi < MSUB; ++mi) {
            ua[mi].h[0] = *(const v8h*)&As[wv][mi][nn][g * 8];
            ua[mi].h[1] = *(const v8h*)&As[wv][mi][nn][16 + g * 8];
        }
        // B (16-bit 32x16): lane holds N = lane%16, K same as A (8*g + h, 16 + 8*g + (h-8))
        #pragma unroll
        for (int nt = 0; nt < 4; ++nt) {
            if (nt < NT) {     // uniform scalar condition: EXEC stays all-1s
                union { v16h v; v8h h[2]; } ub;
                ub.h[0] = *(const v8h*)&Bt[nt][nn][g * 8];
                ub.h[1] = *(const v8h*)&Bt[nt][nn][16 + g * 8];
                v16h b = ub.v;
                #pragma unroll
                for (int mi = 0; mi < MSUB; ++mi) {
                    acc[mi][nt] = __builtin_amdgcn_wmma_f32_16x16x32_f16(
                        false, ua[mi].v, false, b, (short)0, acc[mi][nt], false, false);
                }
            }
        }
        __syncthreads();
    }

    // ----- store C (f32 16x16 layout: VGPR r -> M = 8*g + r, N = lane%16) -----
    if (mbase < Mtot) {
        #pragma unroll
        for (int mi = 0; mi < MSUB; ++mi) {
            #pragma unroll
            for (int r = 0; r < 8; ++r) {
                int vox = mbase + mi * 16 + g * 8 + r;
                if (vox >= Mtot) continue;
                long ovox;
                if (a.up_mode) {
                    int id = vox / (a.IH * a.IW);
                    int rem = vox - id * (a.IH * a.IW);
                    int ih = rem / a.IW;
                    int iw = rem - ih * a.IW;
                    ovox = ((long)(id * a.SD + utd) * a.OH + (ih * a.SH + uth)) * a.OW
                         + (iw * a.SW + utw);
                } else ovox = vox;
                float mo = a.mask_out ? a.mask_out[ovox] : 1.f;
                #pragma unroll
                for (int nt = 0; nt < 4; ++nt) {
                    if (nt >= NT) break;
                    int col = nt * 16 + nn;
                    if (col < a.Cout) {
                        float v = acc[mi][nt][r];
                        v *= mo;
                        if (a.residual) v += a.residual[ovox * (long)a.Cout + col];
                        a.y[ovox * (long)a.out_cstride + a.out_coffset + col] = v;
                    }
                }
            }
        }
    }
}

// ---------------------------------------------------------------------------
// Support kernels
// ---------------------------------------------------------------------------

__global__ void scatter_kernel(const float* pc, float* grid, float* mask, int npts) {
    int p = blockIdx.x * blockDim.x + threadIdx.x;
    if (p >= npts) return;
    const float* row = pc + (size_t)p * 7;
    int d = (int)row[0], h = (int)row[1], w = (int)row[2];
    long v = ((long)d * 64 + h) * 64 + w;
    atomicAdd(&grid[v * 3 + 0], row[4]);
    atomicAdd(&grid[v * 3 + 1], row[5]);
    atomicAdd(&grid[v * 3 + 2], row[6]);
    mask[v] = 1.0f;
}

__global__ void pool_mask_kernel(const float* mi, float* mo,
                                 int OD, int OH, int OW, int kd, int kh, int kw) {
    int idx = blockIdx.x * blockDim.x + threadIdx.x;
    int tot = OD * OH * OW;
    if (idx >= tot) return;
    int od = idx / (OH * OW);
    int rem = idx - od * (OH * OW);
    int oh = rem / OW;
    int ow = rem - oh * OW;
    int IH_ = OH * kh, IW_ = OW * kw;
    float m = 0.f;
    for (int z = 0; z < kd; ++z)
        for (int y = 0; y < kh; ++y)
            for (int x = 0; x < kw; ++x) {
                long iv = ((long)(od * kd + z) * IH_ + (oh * kh + y)) * IW_ + (ow * kw + x);
                m = fmaxf(m, mi[iv]);
            }
    mo[idx] = m;
}

// accum layout: [0..63]=sum(x*m), [64..127]=sum((x*m)^2), [128]=sum(m)
__global__ void bn_stats_kernel(const float* x, const float* mask, float* acc,
                                int C, int cstride, int coffset, long nvox) {
    __shared__ float ssum[64], ssq[64], scnt;
    int t = threadIdx.x;
    if (t < 64) { ssum[t] = 0.f; ssq[t] = 0.f; }
    if (t == 0) scnt = 0.f;
    __syncthreads();
    long total = nvox * C;
    for (long i = (long)blockIdx.x * blockDim.x + t; i < total;
         i += (long)gridDim.x * blockDim.x) {
        long v = i / C;
        int c = (int)(i - v * C);
        float m = mask[v];
        float val = x[v * cstride + coffset + c] * m;
        atomicAdd(&ssum[c], val);
        atomicAdd(&ssq[c], val * val);     // m in {0,1} => (x*m)^2 = x^2*m
        if (c == 0) atomicAdd(&scnt, m);
    }
    __syncthreads();
    if (t < C) { atomicAdd(&acc[t], ssum[t]); atomicAdd(&acc[64 + t], ssq[t]); }
    if (t == 0) atomicAdd(&acc[128], scnt);
}

__global__ void bn_finalize_kernel(const float* acc, const float* bw, const float* bb,
                                   float* scale, float* shift, int C) {
    int c = threadIdx.x;
    if (c >= C) return;
    float n = fmaxf(acc[128], 1.0f);
    float mean = acc[c] / n;
    float var = acc[64 + c] / n - mean * mean;
    float s = bw[c] * rsqrtf(var + 1e-4f);
    scale[c] = s;
    shift[c] = bb[c] - mean * s;
}

__global__ void head_kernel(const float* pc, const float* x, const float* mask,
                            const float* scale, const float* shift,
                            const float* lw, const float* lb, float* out, int npts) {
    int p = blockIdx.x * blockDim.x + threadIdx.x;
    if (p >= npts) return;
    const float* row = pc + (size_t)p * 7;
    int d = (int)row[0], h = (int)row[1], w = (int)row[2];
    long v = ((long)d * 64 + h) * 64 + w;
    float m = mask[v];
    float t = lb[0];
    #pragma unroll
    for (int c = 0; c < 16; ++c) {
        float y = x[v * 16 + c] * scale[c] + shift[c];
        y = fmaxf(y, 0.f) * m;
        t += y * lw[c];
    }
    out[p] = 1.f / (1.f + expf(-t));
}

// ---------------------------------------------------------------------------
// Host orchestration
// ---------------------------------------------------------------------------

static inline int cdiv(int a, int b) { return (a + b - 1) / b; }

extern "C" void kernel_launch(void* const* d_in, const int* in_sizes, int n_in,
                              void* d_out, int out_size, void* d_ws, size_t ws_size,
                              hipStream_t stream) {
    (void)n_in; (void)out_size; (void)ws_size;
    const float* pc = (const float*)d_in[0];
    auto P = [&](int i) { return (const float*)d_in[i]; };
    const int npts = in_sizes[0] / 7;

    // Param flat order (JAX pytree: sorted dict keys, list order, None dropped):
    //  1:bnF.b 2:bnF.w 3:init_conv 4:lin_b 5:lin_w
    //  6,7:L0 bn_down b/w  8,9:L0 bn_up b/w  10:L0 down
    //  11,12:L1 bn_down    13,14:L1 bn_up    15:L1 down
    //  16-21:L2 pre0 (bn1b,bn1w,bn2b,bn2w,c1,c2)  22-27:L2 pre1
    //  28-34:L1 post0 (+nin@34)  35-40:L1 post1
    //  41-46:L1 pre0  47-52:L1 pre1  53:L1 up
    //  54-60:L0 post0 (+nin@60)  61-66:L0 post1
    //  67-72:L0 pre0  73-78:L0 pre1  79:L0 up

    float* ws = (float*)d_ws;
    size_t off = 0;
    auto alloc = [&](size_t n) { float* p = ws + off; off += (n + 63) & ~(size_t)63; return p; };

    const long nv0 = 64L * 64 * 64, nv1 = 64L * 32 * 32, nv2 = 16L * 16 * 16;
    float* grid3 = alloc(nv0 * 3);
    float* mask0 = alloc(nv0);
    float* mask1 = alloc(nv1);
    float* mask2 = alloc(nv2);
    float* bnacc = alloc(129);
    float* bnscl = alloc(64);
    float* bnsft = alloc(64);
    float* xa0 = alloc(nv0 * 16); float* xb0 = alloc(nv0 * 16);
    float* sc0 = alloc(nv0 * 16); float* cat0 = alloc(nv0 * 32);
    float* xa1 = alloc(nv1 * 32); float* xb1 = alloc(nv1 * 32);
    float* sc1 = alloc(nv1 * 32); float* cat1 = alloc(nv1 * 64);
    float* xa2 = alloc(nv2 * 48); float* xb2 = alloc(nv2 * 48);

    hipMemsetAsync(grid3, 0, nv0 * 3 * sizeof(float), stream);
    hipMemsetAsync(mask0, 0, nv0 * sizeof(float), stream);
    scatter_kernel<<<cdiv(npts, 256), 256, 0, stream>>>(pc, grid3, mask0, npts);
    pool_mask_kernel<<<cdiv((int)nv1, 256), 256, 0, stream>>>(mask0, mask1, 64, 32, 32, 1, 2, 2);
    pool_mask_kernel<<<cdiv((int)nv2, 256), 256, 0, stream>>>(mask1, mask2, 16, 16, 16, 4, 2, 2);

    auto launch_conv = [&](ConvArgs& ca) {
        int Mtot = ca.up_mode ? ca.ID * ca.IH * ca.IW : ca.OD * ca.OH * ca.OW;
        int Mtiles = (Mtot + 16 * MSUB - 1) / (16 * MSUB);
        dim3 grd(cdiv(Mtiles, CONV_WAVES), 1, ca.up_mode ? ca.SD * ca.SH * ca.SW : 1);
        dim3 blk(32, CONV_WAVES, 1);
        conv_wmma_kernel<<<grd, blk, 0, stream>>>(ca);
    };

    auto bn = [&](const float* x, int cstr, int coff, const float* mask,
                  const float* bb, const float* bw, int C, long nvox) {
        hipMemsetAsync(bnacc, 0, 129 * sizeof(float), stream);
        long blocks = (nvox * C + 255) / 256;
        if (blocks > 2048) blocks = 2048;
        bn_stats_kernel<<<dim3((unsigned)blocks), 256, 0, stream>>>(x, mask, bnacc, C, cstr, coff, nvox);
        bn_finalize_kernel<<<1, 64, 0, stream>>>(bnacc, bw, bb, bnscl, bnsft, C);
    };

    struct D3 { int D, H, W; };
    D3 d0{64, 64, 64}, d1{64, 32, 32}, d2{16, 16, 16};

    auto conv_std = [&](const float* x, int cstr, int coff, int Cin, D3 di,
                        const float* w, int kd, int kh, int kw,
                        int sd, int sh, int sw, int pd, int ph, int pw,
                        bool use_bn, const float* min_, const float* mout,
                        const float* resid, float* y, int ocstr, int ocoff,
                        int Cout, D3 dout) {
        ConvArgs ca{};
        ca.x = x; ca.w = w; ca.mask_in = min_; ca.mask_out = mout;
        ca.bnscale = use_bn ? bnscl : nullptr;
        ca.bnshift = use_bn ? bnsft : nullptr;
        ca.residual = resid; ca.y = y;
        ca.ID = di.D; ca.IH = di.H; ca.IW = di.W; ca.Cin = Cin;
        ca.in_cstride = cstr; ca.in_coffset = coff;
        ca.OD = dout.D; ca.OH = dout.H; ca.OW = dout.W; ca.Cout = Cout;
        ca.out_cstride = ocstr; ca.out_coffset = ocoff;
        ca.KD = kd; ca.KH = kh; ca.KW = kw;
        ca.SD = sd; ca.SH = sh; ca.SW = sw;
        ca.PD = pd; ca.PH = ph; ca.PW = pw;
        ca.up_mode = 0;
        launch_conv(ca);
    };

    auto conv_up = [&](const float* x, int Cin, D3 di, const float* w,
                       int sd, int sh, int sw, const float* min_, const float* mout,
                       float* y, int ocstr, int ocoff, int Cout, D3 dout) {
        ConvArgs ca{};
        ca.x = x; ca.w = w; ca.mask_in = min_; ca.mask_out = mout;
        ca.bnscale = bnscl; ca.bnshift = bnsft; ca.residual = nullptr; ca.y = y;
        ca.ID = di.D; ca.IH = di.H; ca.IW = di.W; ca.Cin = Cin;
        ca.in_cstride = Cin; ca.in_coffset = 0;
        ca.OD = dout.D; ca.OH = dout.H; ca.OW = dout.W; ca.Cout = Cout;
        ca.out_cstride = ocstr; ca.out_coffset = ocoff;
        ca.KD = sd; ca.KH = sh; ca.KW = sw;
        ca.SD = sd; ca.SH = sh; ca.SW = sw;
        ca.PD = 0; ca.PH = 0; ca.PW = 0;
        ca.up_mode = 1;
        launch_conv(ca);
    };

    auto run_block = [&](const float* xin, int cstr, int coff, int Cin, int Cout,
                         D3 dm, long nv, const float* mask, int kt,
                         const float* ninw,
                         int i_bn1b, int i_bn1w, int i_bn2b, int i_bn2w, int i_c1, int i_c2,
                         float* tmp, float* scb,
                         float* yout, int ocstr, int ocoff) {
        const float* resid;
        if (ninw) {
            conv_std(xin, cstr, coff, Cin, dm, ninw, 1, 1, 1, 1, 1, 1, 0, 0, 0,
                     false, nullptr, nullptr, nullptr, scb, Cout, 0, Cout, dm);
            resid = scb;
        } else {
            resid = xin;   // identity skip: xin standalone with cstride==Cout, coff==0
        }
        bn(xin, cstr, coff, mask, P(i_bn1b), P(i_bn1w), Cin, nv);
        conv_std(xin, cstr, coff, Cin, dm, P(i_c1), kt, 3, 3, 1, 1, 1, kt / 2, 1, 1,
                 true, mask, mask, nullptr, tmp, Cout, 0, Cout, dm);
        bn(tmp, Cout, 0, mask, P(i_bn2b), P(i_bn2w), Cout, nv);
        conv_std(tmp, Cout, 0, Cout, dm, P(i_c2), kt, 3, 3, 1, 1, 1, kt / 2, 1, 1,
                 true, mask, mask, resid, yout, ocstr, ocoff, Cout, dm);
    };

    // ---- forward pass ----
    // init_conv (Cin=3, 1x3x3 SAME) * mask
    conv_std(grid3, 3, 0, 3, d0, P(3), 1, 3, 3, 1, 1, 1, 0, 1, 1,
             false, nullptr, mask0, nullptr, xa0, 16, 0, 16, d0);
    // L0 pre blocks (kt=1); pre1 writes into cat0[:, :16]
    run_block(xa0, 16, 0, 16, 16, d0, nv0, mask0, 1, nullptr, 67, 68, 69, 70, 71, 72,
              xb0, nullptr, xa0, 16, 0);
    run_block(xa0, 16, 0, 16, 16, d0, nv0, mask0, 1, nullptr, 73, 74, 75, 76, 77, 78,
              xb0, nullptr, cat0, 32, 0);
    // down L0: bn_down -> conv (1,2,2) stride (1,2,2) -> * mask1
    bn(cat0, 32, 0, mask0, P(6), P(7), 16, nv0);
    conv_std(cat0, 32, 0, 16, d0, P(10), 1, 2, 2, 1, 2, 2, 0, 0, 0,
             true, mask0, mask1, nullptr, xa1, 32, 0, 32, d1);
    // L1 pre blocks (kt=3); pre1 writes into cat1[:, :32]
    run_block(xa1, 32, 0, 32, 32, d1, nv1, mask1, 3, nullptr, 41, 42, 43, 44, 45, 46,
              xb1, nullptr, xa1, 32, 0);
    run_block(xa1, 32, 0, 32, 32, d1, nv1, mask1, 3, nullptr, 47, 48, 49, 50, 51, 52,
              xb1, nullptr, cat1, 64, 0);
    // down L1: bn_down -> conv (4,2,2) stride (4,2,2) -> * mask2
    bn(cat1, 64, 0, mask1, P(11), P(12), 32, nv1);
    conv_std(cat1, 64, 0, 32, d1, P(15), 4, 2, 2, 4, 2, 2, 0, 0, 0,
             true, mask1, mask2, nullptr, xa2, 48, 0, 48, d2);
    // L2 pre blocks (kt=3)
    run_block(xa2, 48, 0, 48, 48, d2, nv2, mask2, 3, nullptr, 16, 17, 18, 19, 20, 21,
              xb2, nullptr, xa2, 48, 0);
    run_block(xa2, 48, 0, 48, 48, d2, nv2, mask2, 3, nullptr, 22, 23, 24, 25, 26, 27,
              xb2, nullptr, xa2, 48, 0);
    // up to L1: bn_up(L1) -> conv_transpose (4,2,2) -> * mask1 into cat1[:, 32:64]
    bn(xa2, 48, 0, mask2, P(13), P(14), 48, nv2);
    conv_up(xa2, 48, d2, P(53), 4, 2, 2, mask2, mask1, cat1, 64, 32, 32, d1);
    // L1 post blocks
    run_block(cat1, 64, 0, 64, 32, d1, nv1, mask1, 3, P(34), 28, 29, 30, 31, 32, 33,
              xb1, sc1, xa1, 32, 0);
    run_block(xa1, 32, 0, 32, 32, d1, nv1, mask1, 3, nullptr, 35, 36, 37, 38, 39, 40,
              xb1, nullptr, xa1, 32, 0);
    // up to L0: bn_up(L0) -> conv_transpose (1,2,2) -> * mask0 into cat0[:, 16:32]
    bn(xa1, 32, 0, mask1, P(8), P(9), 32, nv1);
    conv_up(xa1, 32, d1, P(79), 1, 2, 2, mask1, mask0, cat0, 32, 16, 16, d0);
    // L0 post blocks (kt=1)
    run_block(cat0, 32, 0, 32, 16, d0, nv0, mask0, 1, P(60), 54, 55, 56, 57, 58, 59,
              xb0, sc0, xa0, 16, 0);
    run_block(xa0, 16, 0, 16, 16, d0, nv0, mask0, 1, nullptr, 61, 62, 63, 64, 65, 66,
              xb0, nullptr, xa0, 16, 0);
    // final bn + gather + linear + sigmoid
    bn(xa0, 16, 0, mask0, P(1), P(2), 16, nv0);
    head_kernel<<<cdiv(npts, 256), 256, 0, stream>>>(pc, xa0, mask0, bnscl, bnsft,
                                                     P(5), P(4), (float*)d_out, npts);
}
